// RandomVectorQuantizer_52656299049000
// MI455X (gfx1250) — compile-verified, matrix operand order checked by reading, not run on hardware
//
#include <hip/hip_runtime.h>
#include <hip/hip_bf16.h>

typedef __attribute__((ext_vector_type(16))) __bf16 v16bf;
typedef __attribute__((ext_vector_type(8)))  __bf16 v8bf;
typedef __attribute__((ext_vector_type(4)))  __bf16 v4bf;
typedef __attribute__((ext_vector_type(8)))  float  v8f;
typedef __attribute__((ext_vector_type(4)))  unsigned int v4u;
typedef __attribute__((ext_vector_type(8)))  int v8i;
typedef __attribute__((ext_vector_type(4)))  int v4i;

// Problem dims (fixed by the reference)
#define DIM_B 4
#define DIM_T 2048
#define DIM_F 1024
#define DIM_D 256
#define DIM_K 8192
#define DIM_M (DIM_B * DIM_T)          // 8192 rows
#define NPB   (DIM_T * DIM_F)          // 2097152 elems per batch

// Argmax kernel tiling
#define CHUNK_ROWS  64                           // codebook rows staged per TDM copy
#define CHUNK_BYTES (CHUNK_ROWS * DIM_D * 2)     // 32768 B (contiguous in cbh)
#define ROW_STRIDE  528                          // 512B row + 16B TDM pad -> bank-conflict-free
#define BUF_BYTES   (CHUNK_ROWS * ROW_STRIDE)    // 33792 B per buffer
#define NSPLIT      8                            // codebook range splits (occupancy)
#define CHUNKS_PER_SPLIT (DIM_K / CHUNK_ROWS / NSPLIT)   // 16

// ---------------------------------------------------------------------------
// WMMA fragment loader: 16-bit A 16x32 layout (ISA 7.12.2).
// lane = half*16 + lr; lane holds row `lr`, chunks at K = kbase + half*8 and
// K = kbase + 16 + half*8 (each 8 bf16 = one b128 load).
// ---------------------------------------------------------------------------
__device__ __forceinline__ v16bf load_frag(const __bf16* row, int kbase, int half) {
    v8bf lo = *(const v8bf*)(row + kbase + half * 8);
    v8bf hi = *(const v8bf*)(row + kbase + 16 + half * 8);
    v16bf r;
#pragma unroll
    for (int i = 0; i < 8; ++i) { r[i] = lo[i]; r[i + 8] = hi[i]; }
    return r;
}

// ---------------------------------------------------------------------------
// Tensor Data Mover: 1-D copy of one codebook chunk (32 KB) into LDS with
// 16B padding inserted every 512B (row stride 528B => LDS banks conflict-free).
// D# layout per CDNA5 ISA 8.3/8.4. Issued once per chunk; tracked by TENSORcnt.
// This toolchain exposes the 6-arg builtin: (g0, g1, g2, g3, g_extra, cpol).
// ---------------------------------------------------------------------------
__device__ __forceinline__ void tdm_load_chunk(unsigned lds_addr, unsigned long long gaddr) {
    v4u g0;
    g0[0] = 1u;                                        // count=1 (valid), no gather
    g0[1] = lds_addr;                                  // lds_addr[63:32]
    g0[2] = (unsigned)(gaddr & 0xFFFFFFFFu);           // global_addr[95:64]
    g0[3] = (unsigned)(gaddr >> 32) | 0x80000000u;     // global_addr[120:96] | type=2
    v8i g1;
    // data_size=1 (2B) <<16 | pad_enable<<20 | pad_interval=6 (128 DW=512B)<<22
    // | pad_amount=3 (4 DW=16B)<<25
    g1[0] = (int)0x07910000;
    g1[1] = (int)0x40000000;   // tensor_dim0[15:0]=16384 in bits[63:48]
    g1[2] = (int)0x00010000;   // tensor_dim0[31:16]=0 | tensor_dim1=1
    g1[3] = (int)0x40000000;   // tensor_dim1[31:16]=0 | tile_dim0=16384
    g1[4] = 1;                 // tile_dim1=1 | tile_dim2=0
    g1[5] = 0x4000;            // tensor_dim0_stride = 16384
    g1[6] = 0;
    g1[7] = 0;
    v4i z4 = {0, 0, 0, 0};     // groups 2/3 unused (<=2D tensor)
    v8i z8 = {0, 0, 0, 0, 0, 0, 0, 0};
    __builtin_amdgcn_tensor_load_to_lds(g0, g1, z4, z4, z8, 0);
}

// ---------------------------------------------------------------------------
// K-init: zero the per-batch stats accumulators in workspace.
// ---------------------------------------------------------------------------
__global__ void rvq_zero_stats(float* stats) {
    if (threadIdx.x < 2 * DIM_B) stats[threadIdx.x] = 0.0f;
}

// ---------------------------------------------------------------------------
// K0: per-batch sum / sum-of-squares. grid = (DIM_B * 32) blocks of 256.
// ---------------------------------------------------------------------------
__global__ void rvq_stats(const float* __restrict__ x, float* __restrict__ stats) {
    const int tid   = threadIdx.x;
    const int b     = blockIdx.x >> 5;
    const int chunk = blockIdx.x & 31;
    const float4* p = (const float4*)(x + (size_t)b * NPB) + (size_t)chunk * 16384;
    float s = 0.0f, q = 0.0f;
#pragma unroll 4
    for (int i = 0; i < 64; ++i) {
        float4 f = p[tid + i * 256];
        s += f.x + f.y + f.z + f.w;
        q += f.x * f.x + f.y * f.y + f.z * f.z + f.w * f.w;
    }
    __shared__ float ls[256], lq[256];
    ls[tid] = s; lq[tid] = q;
    __syncthreads();
    for (int off = 128; off > 0; off >>= 1) {
        if (tid < off) { ls[tid] += ls[tid + off]; lq[tid] += lq[tid + off]; }
        __syncthreads();
    }
    if (tid == 0) {
        atomicAdd(&stats[2 * b + 0], ls[0]);
        atomicAdd(&stats[2 * b + 1], lq[0]);
    }
}

// ---------------------------------------------------------------------------
// K1: normalize x (per-batch mean / unbiased std) and convert to bf16.
// ---------------------------------------------------------------------------
__global__ void rvq_normx(const float* __restrict__ x, const float* __restrict__ stats,
                          __bf16* __restrict__ xh) {
    const size_t g    = (size_t)blockIdx.x * 256 + threadIdx.x;
    const size_t base = g * 16;
    const int b = (int)(base >> 21);                 // NPB == 2^21
    const float s = stats[2 * b + 0], q = stats[2 * b + 1];
    const float n = (float)NPB;
    const float mean = s / n;
    const float var  = (q - s * s / n) / (n - 1.0f); // ddof = 1
    const float rstd = rsqrtf(var);
#pragma unroll
    for (int i = 0; i < 4; ++i) {
        float4 f = *(const float4*)(x + base + i * 4);
        v4bf o;
        o[0] = (__bf16)((f.x - mean) * rstd);
        o[1] = (__bf16)((f.y - mean) * rstd);
        o[2] = (__bf16)((f.z - mean) * rstd);
        o[3] = (__bf16)((f.w - mean) * rstd);
        *(v4bf*)(xh + base + i * 4) = o;
    }
}

// ---------------------------------------------------------------------------
// K2: W [D,F] fp32 -> bf16.
// ---------------------------------------------------------------------------
__global__ void rvq_convw(const float* __restrict__ W, __bf16* __restrict__ wh) {
    const size_t g = (size_t)blockIdx.x * 256 + threadIdx.x;
    float4 f = *(const float4*)(W + g * 4);
    v4bf o;
    o[0] = (__bf16)f.x; o[1] = (__bf16)f.y; o[2] = (__bf16)f.z; o[3] = (__bf16)f.w;
    *(v4bf*)(wh + g * 4) = o;
}

// ---------------------------------------------------------------------------
// K3: GEMM1  v[m,d] = sum_f xh[m,f] * wh[d,f]   (M=8192, N=256, K=1024)
// ---------------------------------------------------------------------------
__global__ void rvq_gemm1(const __bf16* __restrict__ xh, const __bf16* __restrict__ wh,
                          float* __restrict__ v_out, __bf16* __restrict__ vh) {
    const int tid  = threadIdx.x;
    const int wave = tid >> 5, lane = tid & 31;
    const int half = lane >> 4, lr = lane & 15;
    const int mbase = blockIdx.x * 16;

    const __bf16* arow  = xh + (size_t)(mbase + lr) * DIM_F;
    const int nt0 = wave, nt1 = wave + 8;
    const __bf16* brow0 = wh + (size_t)(nt0 * 16 + lr) * DIM_F;
    const __bf16* brow1 = wh + (size_t)(nt1 * 16 + lr) * DIM_F;

    v8f c0 = {}, c1 = {};
#pragma unroll 4
    for (int s = 0; s < DIM_F / 32; ++s) {
        v16bf a  = load_frag(arow,  s * 32, half);
        v16bf b0 = load_frag(brow0, s * 32, half);
        v16bf b1 = load_frag(brow1, s * 32, half);
        c0 = __builtin_amdgcn_wmma_f32_16x16x32_bf16(false, a, false, b0, (short)0, c0, false, false);
        c1 = __builtin_amdgcn_wmma_f32_16x16x32_bf16(false, a, false, b1, (short)0, c1, false, false);
    }
#pragma unroll
    for (int j = 0; j < 8; ++j) {
        const int row  = mbase + j + 8 * half;
        const int col0 = nt0 * 16 + lr;
        const int col1 = nt1 * 16 + lr;
        v_out[(size_t)row * DIM_D + col0] = c0[j];
        v_out[(size_t)row * DIM_D + col1] = c1[j];
        vh[(size_t)row * DIM_D + col0] = (__bf16)c0[j];
        vh[(size_t)row * DIM_D + col1] = (__bf16)c1[j];
    }
}

// ---------------------------------------------------------------------------
// K4: normalize codebook rows, convert to bf16. One wave per row.
// ---------------------------------------------------------------------------
__global__ void rvq_cbnorm(const float* __restrict__ cb, __bf16* __restrict__ cbh) {
    const int row  = blockIdx.x * 8 + (threadIdx.x >> 5);
    const int lane = threadIdx.x & 31;
    const float* r = cb + (size_t)row * DIM_D + lane * 8;
    float a[8];
    float4 f0 = *(const float4*)(r);
    float4 f1 = *(const float4*)(r + 4);
    a[0] = f0.x; a[1] = f0.y; a[2] = f0.z; a[3] = f0.w;
    a[4] = f1.x; a[5] = f1.y; a[6] = f1.z; a[7] = f1.w;
    float sq = 0.0f;
#pragma unroll
    for (int i = 0; i < 8; ++i) sq += a[i] * a[i];
#pragma unroll
    for (int m = 16; m >= 1; m >>= 1) sq += __shfl_xor(sq, m, 32);
    const float rn = 1.0f / fmaxf(sqrtf(sq), 1e-8f);
    v8bf o;
#pragma unroll
    for (int i = 0; i < 8; ++i) o[i] = (__bf16)(a[i] * rn);
    *(v8bf*)(cbh + (size_t)row * DIM_D + lane * 8) = o;
}

// ---------------------------------------------------------------------------
// K5: fused scores GEMM + argmax with TDM-staged codebook tiles.
// grid = 512 blocks of 256 (8 waves). Block = (mblk, split):
//   mblk  = blockIdx.x >> 3 : 128 output rows (wave w owns rows mblk*128+w*16..)
//   split = blockIdx.x & 7  : 1/8 of the codebook (16 chunks of 64 rows)
// Wave 0 double-buffers 32KB codebook chunks into LDS via TENSOR_LOAD_TO_LDS
// (with DMA-inserted padding -> 528B row stride, conflict-free ds_load_b128);
// all 8 waves consume each chunk (8x L2 traffic reduction vs direct loads).
// Per-lane running (max, argmax) in registers; 16-lane shuffle reduce; partial
// results per split go to workspace for the tiny finalize kernel.
// ---------------------------------------------------------------------------
__global__ void rvq_argmax(const __bf16* __restrict__ vh, const __bf16* __restrict__ cbh,
                           float* __restrict__ pv, int* __restrict__ pi) {
    __shared__ __align__(128) unsigned char lds_cb[2 * BUF_BYTES];

    const int tid  = threadIdx.x;
    const int wave = tid >> 5, lane = tid & 31;
    const int half = lane >> 4, lr = lane & 15;
    const int mblk  = blockIdx.x >> 3;
    const int split = blockIdx.x & (NSPLIT - 1);
    const int mw     = mblk * 128 + wave * 16;       // this wave's 16 output rows
    const int chunk0 = split * CHUNKS_PER_SPLIT;

    // Preload the wave's full A panel (16 rows x D=256) into 8 fragments.
    const __bf16* arow = vh + (size_t)(mw + lr) * DIM_D;
    v16bf a[8];
#pragma unroll
    for (int s = 0; s < 8; ++s) a[s] = load_frag(arow, s * 32, half);

    const unsigned lds_base = (unsigned)(uintptr_t)(void*)lds_cb;
    const unsigned long long gbase =
        (unsigned long long)(uintptr_t)cbh + (unsigned long long)chunk0 * CHUNK_BYTES;

    if (wave == 0) tdm_load_chunk(lds_base, gbase);  // prologue: chunk 0 -> buf 0

    float bv[8]; int bi[8];
#pragma unroll
    for (int j = 0; j < 8; ++j) { bv[j] = -3.4e38f; bi[j] = 0; }

    for (int it = 0; it < CHUNKS_PER_SPLIT; ++it) {
        if (wave == 0) __builtin_amdgcn_s_wait_tensorcnt(0);  // chunk `it` landed
        __syncthreads();                                      // LDS visible to all waves
        if (wave == 0 && it + 1 < CHUNKS_PER_SPLIT)           // prefetch next chunk
            tdm_load_chunk(lds_base + ((it + 1) & 1) * BUF_BYTES,
                           gbase + (unsigned long long)(it + 1) * CHUNK_BYTES);

        const unsigned char* buf = lds_cb + (it & 1) * BUF_BYTES;
#pragma unroll
        for (int sub = 0; sub < 4; ++sub) {
            const __bf16* brow = (const __bf16*)(buf + (sub * 16 + lr) * ROW_STRIDE);
            v8f c = {};
#pragma unroll
            for (int s = 0; s < 8; ++s) {
                v16bf b = load_frag(brow, s * 32, half);
                c = __builtin_amdgcn_wmma_f32_16x16x32_bf16(false, a[s], false, b,
                                                            (short)0, c, false, false);
            }
            const int n = (chunk0 + it) * CHUNK_ROWS + sub * 16 + lr;
#pragma unroll
            for (int j = 0; j < 8; ++j)
                if (c[j] > bv[j]) { bv[j] = c[j]; bi[j] = n; }  // strict > keeps first max
        }
    }

    // Reduce across the 16 lanes of each half (same M row, different N).
#pragma unroll
    for (int j = 0; j < 8; ++j) {
        float v = bv[j]; int i = bi[j];
#pragma unroll
        for (int m = 8; m >= 1; m >>= 1) {
            float ov = __shfl_xor(v, m, 32);
            int   oi = __shfl_xor(i, m, 32);
            if (ov > v || (ov == v && oi < i)) { v = ov; i = oi; }
        }
        if (lr == 0) {
            const int row = mw + j + 8 * half;
            pv[row * NSPLIT + split] = v;
            pi[row * NSPLIT + split] = i;
        }
    }
}

// ---------------------------------------------------------------------------
// K6: reduce the 8 per-split candidates per row to the final argmax.
// ---------------------------------------------------------------------------
__global__ void rvq_finalize(const float* __restrict__ pv, const int* __restrict__ pi,
                             float* __restrict__ t_out) {
    const int row = blockIdx.x * 256 + threadIdx.x;
    float v = pv[row * NSPLIT]; int i = pi[row * NSPLIT];
#pragma unroll
    for (int s = 1; s < NSPLIT; ++s) {
        float ov = pv[row * NSPLIT + s]; int oi = pi[row * NSPLIT + s];
        if (ov > v || (ov == v && oi < i)) { v = ov; i = oi; }
    }
    t_out[row] = (float)i;
}

// ---------------------------------------------------------------------------
// Host-side launcher
// ---------------------------------------------------------------------------
extern "C" void kernel_launch(void* const* d_in, const int* in_sizes, int n_in,
                              void* d_out, int out_size, void* d_ws, size_t ws_size,
                              hipStream_t stream) {
    const float* x  = (const float*)d_in[0];   // [B,T,F]
    const float* W  = (const float*)d_in[1];   // [D,F]
    const float* cb = (const float*)d_in[2];   // [K,D]

    float* v_out = (float*)d_out;                         // [M,D] fp32
    float* t_out = v_out + (size_t)DIM_M * DIM_D;         // [M] indices (as float)

    // Workspace layout (256-byte aligned regions)
    char* ws = (char*)d_ws;
    float*  stats = (float*)ws;                                         // 8 f32
    __bf16* xh    = (__bf16*)(ws + 256);                                // M x F  (16 MB)
    __bf16* wh    = (__bf16*)((char*)xh + (size_t)DIM_M * DIM_F * 2);   // D x F  (0.5 MB)
    __bf16* vh    = (__bf16*)((char*)wh + (size_t)DIM_D * DIM_F * 2);   // M x D  (4 MB)
    __bf16* cbh   = (__bf16*)((char*)vh + (size_t)DIM_M * DIM_D * 2);   // K x D  (4 MB)
    float*  pv    = (float*) ((char*)cbh + (size_t)DIM_K * DIM_D * 2);  // M x 8  (256 KB)
    int*    pi    = (int*)   ((char*)pv  + (size_t)DIM_M * NSPLIT * 4); // M x 8  (256 KB)

    rvq_zero_stats<<<1, 32, 0, stream>>>(stats);
    rvq_stats<<<DIM_B * 32, 256, 0, stream>>>(x, stats);
    rvq_normx<<<2048, 256, 0, stream>>>(x, stats, xh);
    rvq_convw<<<256, 256, 0, stream>>>(W, wh);
    rvq_gemm1<<<DIM_M / 16, 256, 0, stream>>>(xh, wh, v_out, vh);
    rvq_cbnorm<<<DIM_K / 8, 256, 0, stream>>>(cb, cbh);
    rvq_argmax<<<(DIM_M / 128) * NSPLIT, 256, 0, stream>>>(vh, cbh, pv, pi);
    rvq_finalize<<<DIM_M / 256, 256, 0, stream>>>(pv, pi, t_out);
}
